// GQAMultiHeadAttention_20495583936779
// MI455X (gfx1250) — matrix operand order repaired and checked by reading, not compiled, not run on hardware
//
#include <hip/hip_runtime.h>
#include <hip/hip_bf16.h>
#include <math.h>

typedef __attribute__((ext_vector_type(16))) _Float16 v16h;
typedef __attribute__((ext_vector_type(8)))  _Float16 h16x8;
typedef __attribute__((ext_vector_type(8)))  float    v8f;
typedef __attribute__((ext_vector_type(4)))  float    f32x4;
typedef __attribute__((ext_vector_type(4)))  _Float16 h16x4;
typedef __attribute__((ext_vector_type(4)))  unsigned int u32x4;
typedef __attribute__((ext_vector_type(8)))  int      i32x8;
typedef __attribute__((ext_vector_type(4)))  int      i32x4;

// Problem constants
constexpr int Bc  = 2;
constexpr int Tc  = 2048;
constexpr int Dc  = 1024;
constexpr int Hc  = 16;
constexpr int KVc = 4;
constexpr int HDc = 64;
constexpr int NQKV = Hc * HDc + 2 * KVc * HDc; // 1536
constexpr int BT  = Bc * Tc;                   // 4096

__device__ inline v8f wmma32(v16h a, v16h b, v8f c) {
  // D = A(16x32 f16) * B(32x16 f16) + C(16x16 f32)
  return __builtin_amdgcn_wmma_f32_16x16x32_f16(false, a, false, b, (short)0, c,
                                                false, false);
}

// ---- fragment gathers (per CDNA5 ISA 7.12.2 VGPR layouts, wave32) ----

// A operand, source row-major [M][K] in LDS.
__device__ inline v16h frag_a(const _Float16* base, int ld, int lane) {
  const int m = lane & 15, hf = lane >> 4;
  v16h f;
#pragma unroll
  for (int e = 0; e < 16; ++e) {
    const int g = e >> 1, p = e & 1;
    const int kb = (g < 4) ? (g * 2) : (16 + (g - 4) * 2);
    f[e] = base[m * ld + kb + hf * 8 + p];
  }
  return f;
}

// A operand gathered straight from (16B-aligned) row-major global memory:
// the swizzle is two contiguous 8-half runs per lane -> two b128 loads.
__device__ inline v16h frag_a_direct(const _Float16* base, int ld, int k0, int lane) {
  const int m = lane & 15, hf = lane >> 4;
  const h16x8 lo = *(const h16x8*)&base[(size_t)m * ld + k0 + hf * 8];
  const h16x8 hi = *(const h16x8*)&base[(size_t)m * ld + k0 + 16 + hf * 8];
  v16h f;
#pragma unroll
  for (int e = 0; e < 8; ++e) { f[e] = lo[e]; f[8 + e] = hi[e]; }
  return f;
}

// B operand, source [N][K] row-major. lane&15 = N, K = (e&7) + 8*(lane>>4) + 16*(e>>3).
__device__ inline v16h frag_b_nmajor(const _Float16* base, int ld, int lane) {
  const int n = lane & 15, hf = lane >> 4;
  v16h f;
#pragma unroll
  for (int e = 0; e < 16; ++e) f[e] = base[n * ld + hf * 8 + (e & 7) + (e >> 3) * 16];
  return f;
}

// B operand, source [K][N] row-major (V tile: rows = keys).
__device__ inline v16h frag_b_kmajor(const _Float16* base, int ld, int lane) {
  const int n = lane & 15, hf = lane >> 4;
  v16h f;
#pragma unroll
  for (int e = 0; e < 16; ++e) f[e] = base[(hf * 8 + (e & 7) + (e >> 3) * 16) * ld + n];
  return f;
}

// ---- Tensor Data Mover: load a 32x64 f16 tile (row-major, stride 64 elems)
// into LDS with 16B padding every 128B (-> 72-half LDS row stride). ----
__device__ inline void tdm_load_tile_32x64_f16(unsigned lds_addr, const void* gptr) {
  const unsigned long long ga = (unsigned long long)gptr;
  u32x4 g0;
  g0[0] = 1u;                                        // count=1, user mode
  g0[1] = lds_addr;                                  // LDS byte address
  g0[2] = (unsigned)(ga & 0xffffffffu);              // global addr [31:0]
  g0[3] = (unsigned)((ga >> 32) & 0x01ffffffu)       // global addr [56:32]
          | (2u << 30);                              // type=2 (image)
  i32x8 g1;
  g1[0] = (1 << 16)    // data_size = 1 -> 2 bytes
        | (1 << 20)    // pad_enable
        | (4 << 22)    // pad_interval: code 4 -> every 32 DWORDs (128B)
        | (3 << 25);   // pad_amount:   code 3 -> 4 DWORDs (16B)
  g1[1] = (64 << 16);        // tensor_dim0 = 64 (bits 79:48)
  g1[2] = (int)(2048u << 16);// tensor_dim1 = 2048 (bits 111:80)
  g1[3] = (64 << 16);        // tile_dim0 = 64 (bits 127:112)
  g1[4] = 32;                // tile_dim1 = 32 rows
  g1[5] = 64;                // tensor_dim0_stride = 64 elements
  g1[6] = 0;
  g1[7] = 0;
  i32x4 z = {0, 0, 0, 0};
  asm volatile("" ::: "memory");
#if __has_include(<hip/amd_detail/amd_gfx1250_TDM.h>)
  i32x8 z8 = {0, 0, 0, 0, 0, 0, 0, 0};
  __builtin_amdgcn_tensor_load_to_lds(g0, g1, z, z, z8, 0);
#else
  __builtin_amdgcn_tensor_load_to_lds(g0, g1, z, z, 0);
#endif
  asm volatile("" ::: "memory");
}

// ================= Kernel 1: fused QKV projection =================
__global__ __launch_bounds__(128) void qkv_gemm_kernel(
    const float* __restrict__ x, const float* __restrict__ wq,
    const float* __restrict__ wk, const float* __restrict__ wv,
    float* __restrict__ qkv) {
  __shared__ __align__(16) _Float16 As[16][40];
  __shared__ __align__(16) _Float16 Bs[64][40];
  const int tid = threadIdx.x, wave = tid >> 5, lane = tid & 31;
  const int m0 = blockIdx.y * 16;
  const int n0 = blockIdx.x * 64;
  const int nw = n0 + wave * 16;
  v8f acc = {};
  for (int kk = 0; kk < Dc; kk += 32) {
    {
      const int r = tid >> 3, c = (tid & 7) * 4;
      const f32x4 v = *(const f32x4*)&x[(size_t)(m0 + r) * Dc + kk + c];
      h16x4 hv;
      hv[0] = (_Float16)v[0]; hv[1] = (_Float16)v[1];
      hv[2] = (_Float16)v[2]; hv[3] = (_Float16)v[3];
      *(h16x4*)&As[r][c] = hv;
    }
#pragma unroll
    for (int it = 0; it < 4; ++it) {
      const int idx = tid + it * 128;
      const int r = idx >> 3, c = (idx & 7) * 4;
      const int n = n0 + r;
      const float* wrow;
      int nr;
      if (n < Hc * HDc)                  { wrow = wq; nr = n; }
      else if (n < (Hc + KVc) * HDc)     { wrow = wk; nr = n - Hc * HDc; }
      else                               { wrow = wv; nr = n - (Hc + KVc) * HDc; }
      const f32x4 v = *(const f32x4*)&wrow[(size_t)nr * Dc + kk + c];
      h16x4 hv;
      hv[0] = (_Float16)v[0]; hv[1] = (_Float16)v[1];
      hv[2] = (_Float16)v[2]; hv[3] = (_Float16)v[3];
      *(h16x4*)&Bs[r][c] = hv;
    }
    __syncthreads();
    const v16h a = frag_a(&As[0][0], 40, lane);
    const v16h b = frag_b_nmajor(&Bs[wave * 16][0], 40, lane);
    acc = wmma32(a, b, acc);
    __syncthreads();
  }
  const int hf = lane >> 4, col = lane & 15;
#pragma unroll
  for (int r = 0; r < 8; ++r) {
    const int m = m0 + r + 8 * hf;
    qkv[(size_t)m * NQKV + nw + col] = acc[r];
  }
}

// ================= Kernel 2: RoPE + repack to f16 =================
__global__ void rope_pack_kernel(const float* __restrict__ qkv,
                                 _Float16* __restrict__ Qh,
                                 _Float16* __restrict__ Kh,
                                 _Float16* __restrict__ Vh) {
  constexpr int PER_BT = Hc * 32 + KVc * 32 + KVc * 64; // 896
  constexpr float NEG_LOG2_BASE_OVER_32 = -0.41524101186092f; // -log2(10000)/32
  int idx = blockIdx.x * blockDim.x + threadIdx.x;
  if (idx >= BT * PER_BT) return;
  const int bt = idx / PER_BT;
  int w = idx - bt * PER_BT;
  const int b = bt / Tc, t = bt - b * Tc;
  const float* row = qkv + (size_t)bt * NQKV;
  if (w < Hc * 32) {
    const int h = w >> 5, i = w & 31;
    const float ang = (float)t * exp2f((float)i * NEG_LOG2_BASE_OVER_32);
    const float c = __cosf(ang), s = __sinf(ang);
    const float x1 = row[h * HDc + i], x2 = row[h * HDc + 32 + i];
    const size_t o = ((size_t)(b * Hc + h) * Tc + t) * HDc;
    Qh[o + i]      = (_Float16)(x1 * c - x2 * s);
    Qh[o + 32 + i] = (_Float16)(x2 * c + x1 * s);
  } else if (w < Hc * 32 + KVc * 32) {
    w -= Hc * 32;
    const int h = w >> 5, i = w & 31;
    const float ang = (float)t * exp2f((float)i * NEG_LOG2_BASE_OVER_32);
    const float c = __cosf(ang), s = __sinf(ang);
    const float x1 = row[Hc * HDc + h * HDc + i];
    const float x2 = row[Hc * HDc + h * HDc + 32 + i];
    const size_t o = ((size_t)(b * KVc + h) * Tc + t) * HDc;
    Kh[o + i]      = (_Float16)(x1 * c - x2 * s);
    Kh[o + 32 + i] = (_Float16)(x2 * c + x1 * s);
  } else {
    w -= Hc * 32 + KVc * 32;
    const int h = w >> 6, i = w & 63;
    const size_t o = ((size_t)(b * KVc + h) * Tc + t) * HDc;
    Vh[o + i] = (_Float16)row[(Hc + KVc) * HDc + h * HDc + i];
  }
}

// ================= Kernel 3: causal flash attention =================
// 4 waves/block = 64 query rows sharing double-buffered TDM-staged K/V tiles.
// Wave 0 DMAs tile jb+1 while all waves compute tile jb (s_wait_tensorcnt 2).
__global__ __launch_bounds__(128) void attn_kernel(const _Float16* __restrict__ Qh,
                                                   const _Float16* __restrict__ Kh,
                                                   const _Float16* __restrict__ Vh,
                                                   _Float16* __restrict__ Ah) {
  __shared__ __align__(16) _Float16 Ks[2][32][72];
  __shared__ __align__(16) _Float16 Vs[2][32][72];
  __shared__ __align__(16) _Float16 Ps[4][16][40];
  const int tid = threadIdx.x, wave = tid >> 5, lane = tid & 31;
  const int h = blockIdx.y, b = blockIdx.z;
  const int q0b = blockIdx.x * 64;        // block's 64 query rows
  const int q0w = q0b + wave * 16;        // this wave's 16 query rows
  const int kvh = h / (Hc / KVc);
  const _Float16* Qg = Qh + ((size_t)(b * Hc + h) * Tc + q0w) * HDc;
  const _Float16* Kg = Kh + ((size_t)(b * KVc + kvh) * Tc) * HDc;
  const _Float16* Vg = Vh + ((size_t)(b * KVc + kvh) * Tc) * HDc;
  const unsigned ldsK[2] = {(unsigned)(size_t)&Ks[0][0][0], (unsigned)(size_t)&Ks[1][0][0]};
  const unsigned ldsV[2] = {(unsigned)(size_t)&Vs[0][0][0], (unsigned)(size_t)&Vs[1][0][0]};

  // Q fragments direct from global (two b128 loads each)
  const v16h aq0 = frag_a_direct(Qg, HDc, 0, lane);
  const v16h aq1 = frag_a_direct(Qg, HDc, 32, lane);

  v8f acc0 = {}, acc1 = {}, acc2 = {}, acc3 = {};
  float mrow[8], lrow[8];
#pragma unroll
  for (int r = 0; r < 8; ++r) { mrow[r] = -1e30f; lrow[r] = 0.0f; }

  const int hf = lane >> 4, col = lane & 15;
  const int nblk = (q0b + 64) / 32;       // q0b+64 is a multiple of 32
  const float scale = 0.125f;             // 1/sqrt(64)

  if (wave == 0) {                        // prologue: DMA tile 0
    tdm_load_tile_32x64_f16(ldsK[0], (const void*)Kg);
    tdm_load_tile_32x64_f16(ldsV[0], (const void*)Vg);
  }

  for (int jb = 0; jb < nblk; ++jb) {
    const int j0 = jb * 32;
    const int cur = jb & 1;
    if (wave == 0) {
      if (jb + 1 < nblk) {                // pipeline: DMA next tile, wait current
        tdm_load_tile_32x64_f16(ldsK[cur ^ 1], (const void*)(Kg + (size_t)(j0 + 32) * HDc));
        tdm_load_tile_32x64_f16(ldsV[cur ^ 1], (const void*)(Vg + (size_t)(j0 + 32) * HDc));
        __builtin_amdgcn_s_wait_tensorcnt(2);
      } else {
        __builtin_amdgcn_s_wait_tensorcnt(0);
      }
    }
    __syncthreads();                      // current K/V tile visible to all waves

    if (j0 <= q0w + 15) {                 // wave-uniform causal skip
      v8f s0 = {}, s1 = {};
      s0 = wmma32(aq0, frag_b_nmajor(&Ks[cur][0][0], 72, lane), s0);
      s0 = wmma32(aq1, frag_b_nmajor(&Ks[cur][0][32], 72, lane), s0);
      s1 = wmma32(aq0, frag_b_nmajor(&Ks[cur][16][0], 72, lane), s1);
      s1 = wmma32(aq1, frag_b_nmajor(&Ks[cur][16][32], 72, lane), s1);

      float mnew[8], alpha[8];
#pragma unroll
      for (int r = 0; r < 8; ++r) {
        const int q = q0w + r + 8 * hf;
        const int jA = j0 + col, jB = j0 + 16 + col;
        float a = (jA > q) ? -1e30f : s0[r] * scale;
        float c = (jB > q) ? -1e30f : s1[r] * scale;
        s0[r] = a; s1[r] = c;
        float mx = fmaxf(a, c);
        mx = fmaxf(mx, __shfl_xor(mx, 1));
        mx = fmaxf(mx, __shfl_xor(mx, 2));
        mx = fmaxf(mx, __shfl_xor(mx, 4));
        mx = fmaxf(mx, __shfl_xor(mx, 8));
        mnew[r] = fmaxf(mrow[r], mx);
      }
#pragma unroll
      for (int r = 0; r < 8; ++r) {
        alpha[r] = __expf(mrow[r] - mnew[r]);
        const float p0 = __expf(s0[r] - mnew[r]);
        const float p1 = __expf(s1[r] - mnew[r]);
        s0[r] = p0; s1[r] = p1;
        float sum = p0 + p1;
        sum += __shfl_xor(sum, 1);
        sum += __shfl_xor(sum, 2);
        sum += __shfl_xor(sum, 4);
        sum += __shfl_xor(sum, 8);
        lrow[r] = lrow[r] * alpha[r] + sum;
        mrow[r] = mnew[r];
        acc0[r] *= alpha[r]; acc1[r] *= alpha[r];
        acc2[r] *= alpha[r]; acc3[r] *= alpha[r];
      }
#pragma unroll
      for (int r = 0; r < 8; ++r) {       // Ps is per-wave private
        Ps[wave][r + 8 * hf][col]      = (_Float16)s0[r];
        Ps[wave][r + 8 * hf][16 + col] = (_Float16)s1[r];
      }
      const v16h ap = frag_a(&Ps[wave][0][0], 40, lane);
      acc0 = wmma32(ap, frag_b_kmajor(&Vs[cur][0][0],  72, lane), acc0);
      acc1 = wmma32(ap, frag_b_kmajor(&Vs[cur][0][16], 72, lane), acc1);
      acc2 = wmma32(ap, frag_b_kmajor(&Vs[cur][0][32], 72, lane), acc2);
      acc3 = wmma32(ap, frag_b_kmajor(&Vs[cur][0][48], 72, lane), acc3);
    }
    __syncthreads();                      // all reads of buf[cur] done
  }

#pragma unroll
  for (int r = 0; r < 8; ++r) {
    const float inv = 1.0f / lrow[r];
    const int q = q0w + r + 8 * hf;
    const size_t o = ((size_t)(b * Tc + q) * Hc + h) * HDc;
    Ah[o + col]      = (_Float16)(acc0[r] * inv);
    Ah[o + 16 + col] = (_Float16)(acc1[r] * inv);
    Ah[o + 32 + col] = (_Float16)(acc2[r] * inv);
    Ah[o + 48 + col] = (_Float16)(acc3[r] * inv);
  }
}

// ================= Kernel 4: output projection =================
__global__ __launch_bounds__(128) void out_gemm_kernel(const _Float16* __restrict__ Ah,
                                                       const float* __restrict__ wo,
                                                       float* __restrict__ out) {
  __shared__ __align__(16) _Float16 As[16][40];
  __shared__ __align__(16) _Float16 Bs[64][40];
  const int tid = threadIdx.x, wave = tid >> 5, lane = tid & 31;
  const int m0 = blockIdx.y * 16;
  const int n0 = blockIdx.x * 64;
  const int nw = n0 + wave * 16;
  v8f acc = {};
  for (int kk = 0; kk < Dc; kk += 32) {
    {
      const int r = tid >> 3, c = (tid & 7) * 4;
      *(h16x4*)&As[r][c] = *(const h16x4*)&Ah[(size_t)(m0 + r) * Dc + kk + c];
    }
#pragma unroll
    for (int it = 0; it < 4; ++it) {
      const int idx = tid + it * 128;
      const int r = idx >> 3, c = (idx & 7) * 4;
      const f32x4 v = *(const f32x4*)&wo[(size_t)(n0 + r) * Dc + kk + c];
      h16x4 hv;
      hv[0] = (_Float16)v[0]; hv[1] = (_Float16)v[1];
      hv[2] = (_Float16)v[2]; hv[3] = (_Float16)v[3];
      *(h16x4*)&Bs[r][c] = hv;
    }
    __syncthreads();
    const v16h a = frag_a(&As[0][0], 40, lane);
    const v16h b = frag_b_nmajor(&Bs[wave * 16][0], 40, lane);
    acc = wmma32(a, b, acc);
    __syncthreads();
  }
  const int hf = lane >> 4, col = lane & 15;
#pragma unroll
  for (int r = 0; r < 8; ++r)
    out[(size_t)(m0 + r + 8 * hf) * Dc + nw + col] = acc[r];
}

extern "C" void kernel_launch(void* const* d_in, const int* in_sizes, int n_in,
                              void* d_out, int out_size, void* d_ws, size_t ws_size,
                              hipStream_t stream) {
  const float* x  = (const float*)d_in[0];
  const float* wq = (const float*)d_in[1];
  const float* wk = (const float*)d_in[2];
  const float* wv = (const float*)d_in[3];
  const float* wo = (const float*)d_in[4];
  float* out = (float*)d_out;

  float* qkv = (float*)d_ws;                                   // [4096 x 1536] f32
  _Float16* Qh = (_Float16*)(qkv + (size_t)BT * NQKV);         // [B,H,T,64]  f16
  _Float16* Kh = Qh + (size_t)Bc * Hc * Tc * HDc;              // [B,KV,T,64] f16
  _Float16* Vh = Kh + (size_t)Bc * KVc * Tc * HDc;             // [B,KV,T,64] f16
  _Float16* Ah = Vh + (size_t)Bc * KVc * Tc * HDc;             // [B,T,H,64]  f16

  qkv_gemm_kernel<<<dim3(NQKV / 64, BT / 16), 128, 0, stream>>>(x, wq, wk, wv, qkv);

  constexpr int PER_BT = Hc * 32 + KVc * 32 + KVc * 64;
  const int total = BT * PER_BT;
  rope_pack_kernel<<<(total + 255) / 256, 256, 0, stream>>>(qkv, Qh, Kh, Vh);

  attn_kernel<<<dim3(Tc / 64, Hc, Bc), 128, 0, stream>>>(Qh, Kh, Vh, Ah);

  out_gemm_kernel<<<dim3(Dc / 64, BT / 16), 128, 0, stream>>>(Ah, wo, out);
}
